// LinearPerBlockQuant_13520557047882
// MI455X (gfx1250) — matrix operand order repaired and checked by reading, not compiled, not run on hardware
//
#include <hip/hip_runtime.h>
#include <hip/hip_bf16.h>

// ---------- types ----------
typedef __attribute__((ext_vector_type(16))) _Float16 v16h;
typedef __attribute__((ext_vector_type(8)))  _Float16 v8h;
typedef __attribute__((ext_vector_type(4)))  _Float16 v4h;
typedef __attribute__((ext_vector_type(8)))  float    v8f;
typedef int v4i_vs __attribute__((vector_size(16)));

#define QMAX 255.0f

#define AS1 __attribute__((address_space(1)))
#define AS3 __attribute__((address_space(3)))

#if __has_builtin(__builtin_amdgcn_global_load_async_to_lds_b128)
#define HAVE_ASYNC_LDS 1
#else
#define HAVE_ASYNC_LDS 0
#endif

// 16-byte global -> LDS copy (async if available).
// Builtin signature (from compiler diagnostic): (v4i AS1*, v4i AS3*, Ii off, Ii cpol).
// Generic->AS casts go through integers: flat->global is the same 64-bit address,
// flat->LDS is the low 32 bits (aperture_hi:offset form) — matches the backend's
// addrspacecast lowering.
__device__ __forceinline__ void cp16_g2s(const _Float16* g, _Float16* s) {
#if HAVE_ASYNC_LDS
    __builtin_amdgcn_global_load_async_to_lds_b128(
        (AS1 v4i_vs*)(uintptr_t)g,
        (AS3 v4i_vs*)(uint32_t)(uintptr_t)s,
        /*offset=*/0, /*cpol=*/0);
#else
    *reinterpret_cast<v8h*>(s) = *reinterpret_cast<const v8h*>(g);
#endif
}

__device__ __forceinline__ void wait_async_copies() {
#if HAVE_ASYNC_LDS
#if __has_builtin(__builtin_amdgcn_s_wait_asynccnt)
    __builtin_amdgcn_s_wait_asynccnt(0);
#else
    asm volatile("s_wait_asynccnt 0x0" ::: "memory");
#endif
#endif
}

// ---------- fake-quant helpers ----------
__device__ __forceinline__ float fq1(float v, float s, float z) {
    float q = rintf(v / s + z);          // jnp.round == round-half-even == rint
    q = fminf(fmaxf(q, 0.0f), QMAX);
    return (q - z) * s;
}

// x: [B, IN] f32; per-column-block (nb blocks of 4) scales/zeros shared by all rows.
__global__ void fq_x_kernel(const float* __restrict__ x,
                            const float* __restrict__ a_scales,
                            const float* __restrict__ a_zeros,
                            _Float16* __restrict__ xq,
                            int total_blocks, int nb) {
    int i = blockIdx.x * blockDim.x + threadIdx.x;
    if (i >= total_blocks) return;
    int b = i % nb;
    float s = a_scales[b], z = a_zeros[b];
    float4 v = reinterpret_cast<const float4*>(x)[i];
    v4h o;
    o.x = (_Float16)fq1(v.x, s, z);
    o.y = (_Float16)fq1(v.y, s, z);
    o.z = (_Float16)fq1(v.z, s, z);
    o.w = (_Float16)fq1(v.w, s, z);
    reinterpret_cast<v4h*>(xq)[i] = o;
}

// w: [OUT, IN] f32; scales/zeros are [OUT, nb] flat -> index == flat block index.
__global__ void fq_w_kernel(const float* __restrict__ w,
                            const float* __restrict__ w_scales,
                            const float* __restrict__ w_zeros,
                            _Float16* __restrict__ wq,
                            int total_blocks) {
    int i = blockIdx.x * blockDim.x + threadIdx.x;
    if (i >= total_blocks) return;
    float s = w_scales[i], z = w_zeros[i];
    float4 v = reinterpret_cast<const float4*>(w)[i];
    v4h o;
    o.x = (_Float16)fq1(v.x, s, z);
    o.y = (_Float16)fq1(v.y, s, z);
    o.z = (_Float16)fq1(v.z, s, z);
    o.w = (_Float16)fq1(v.w, s, z);
    reinterpret_cast<v4h*>(wq)[i] = o;
}

// ---------- WMMA GEMM with double-buffered async LDS staging ----------
// C[B, OUT] = xq[B, IN] @ wq[OUT, IN]^T + bias
// Block: 256 threads (8 wave32). Block tile 128(M) x 128(N). K-stage = 64.
// Waves 2(M) x 4(N); each wave computes a 64x32 tile = 4x2 WMMA accumulators.
// LDS tiles padded to row stride 72 halfs (144 B) -> 16 rows hit 16 disjoint
// 4-bank groups, so per-halfwave b128 fragment reads are bank-conflict-free.

#define KSTAGE 64
#define LDS_STRIDE 72          // halfs per LDS tile row (64 + 8 pad)

__global__ __launch_bounds__(256)
void wmma_gemm_kernel(const _Float16* __restrict__ A,   // xq [B, IN]
                      const _Float16* __restrict__ W,   // wq [OUT, IN]
                      const float*    __restrict__ bias,
                      float* __restrict__ C,
                      int IN, int OUT) {
    __shared__ _Float16 sA[2][128 * LDS_STRIDE];
    __shared__ _Float16 sB[2][128 * LDS_STRIDE];

    const int tid  = threadIdx.x;
    const int lane = tid & 31;
    const int wave = tid >> 5;
    const int tm   = lane & 15;
    const int th   = lane >> 4;

    const int wm = wave & 1;              // 2 waves along M
    const int wn = wave >> 1;             // 4 waves along N
    const int m0 = blockIdx.y * 128;
    const int n0 = blockIdx.x * 128;

    const _Float16* gA = A + (size_t)m0 * IN;   // 128 rows of xq
    const _Float16* gB = W + (size_t)n0 * IN;   // 128 rows of wq

    // Per-thread copy assignment: tile = 128 rows x 8 chunks(16B) = 1024 chunks,
    // 4 chunks per thread; consecutive tids cover consecutive 16B of a row.
    const int c_row = tid >> 3;           // 0..31 (then +32 per iteration)
    const int c_kc  = (tid & 7) * 8;      // half offset within stage

    v8f acc[4][2] = {};

    const int nStages = IN / KSTAGE;

    // Prime stage 0.
#pragma unroll
    for (int r = 0; r < 4; ++r) {
        int row = c_row + r * 32;
        cp16_g2s(gA + (size_t)row * IN + c_kc, &sA[0][row * LDS_STRIDE + c_kc]);
        cp16_g2s(gB + (size_t)row * IN + c_kc, &sB[0][row * LDS_STRIDE + c_kc]);
    }
    wait_async_copies();
    __syncthreads();

    int buf = 0;
    for (int s = 0; s < nStages; ++s) {
        // Issue async copies for the next stage into the other buffer.
        if (s + 1 < nStages) {
            const int k0 = (s + 1) * KSTAGE;
#pragma unroll
            for (int r = 0; r < 4; ++r) {
                int row = c_row + r * 32;
                cp16_g2s(gA + (size_t)row * IN + k0 + c_kc,
                         &sA[buf ^ 1][row * LDS_STRIDE + c_kc]);
                cp16_g2s(gB + (size_t)row * IN + k0 + c_kc,
                         &sB[buf ^ 1][row * LDS_STRIDE + c_kc]);
            }
        }

        // Compute current stage from LDS: two K=32 WMMA steps.
        const _Float16* tA = &sA[buf][0];
        const _Float16* tB = &sB[buf][0];
#pragma unroll
        for (int kk = 0; kk < KSTAGE; kk += 32) {
            v16h a[4];
#pragma unroll
            for (int i = 0; i < 4; ++i) {
                const _Float16* p =
                    tA + (wm * 64 + i * 16 + tm) * LDS_STRIDE + kk;
                v8h lo = *reinterpret_cast<const v8h*>(p + th * 8);
                v8h hi = *reinterpret_cast<const v8h*>(p + 16 + th * 8);
                a[i] = __builtin_shufflevector(lo, hi,
                        0, 1, 2, 3, 4, 5, 6, 7, 8, 9, 10, 11, 12, 13, 14, 15);
            }
            v16h b[2];
#pragma unroll
            for (int j = 0; j < 2; ++j) {
                const _Float16* p =
                    tB + (wn * 32 + j * 16 + tm) * LDS_STRIDE + kk;
                v8h lo = *reinterpret_cast<const v8h*>(p + th * 8);
                v8h hi = *reinterpret_cast<const v8h*>(p + 16 + th * 8);
                b[j] = __builtin_shufflevector(lo, hi,
                        0, 1, 2, 3, 4, 5, 6, 7, 8, 9, 10, 11, 12, 13, 14, 15);
            }
#pragma unroll
            for (int i = 0; i < 4; ++i)
#pragma unroll
                for (int j = 0; j < 2; ++j)
                    acc[i][j] = __builtin_amdgcn_wmma_f32_16x16x32_f16(
                        false, a[i], false, b[j],
                        (short)0, acc[i][j], false, false);
        }

        // Next-stage copies must have landed before anyone reads buf^1.
        wait_async_copies();
        __syncthreads();
        buf ^= 1;
    }

    // Store: lane tm = column, VGPR r = row (r + th*8) inside each 16x16 tile.
#pragma unroll
    for (int j = 0; j < 2; ++j) {
        const int col = n0 + wn * 32 + j * 16 + tm;
        const float bb = bias[col];
#pragma unroll
        for (int i = 0; i < 4; ++i) {
            const int rowBase = m0 + wm * 64 + i * 16 + th * 8;
#pragma unroll
            for (int r = 0; r < 8; ++r)
                C[(size_t)(rowBase + r) * OUT + col] = acc[i][j][r] + bb;
        }
    }
}

// ---------- launch ----------
extern "C" void kernel_launch(void* const* d_in, const int* in_sizes, int n_in,
                              void* d_out, int out_size, void* d_ws, size_t ws_size,
                              hipStream_t stream) {
    const float* x        = (const float*)d_in[0];
    const float* weight   = (const float*)d_in[1];
    const float* bias     = (const float*)d_in[2];
    const float* w_scales = (const float*)d_in[3];
    const float* w_zeros  = (const float*)d_in[4];
    const float* a_scales = (const float*)d_in[5];
    const float* a_zeros  = (const float*)d_in[6];
    float* out = (float*)d_out;

    const int nb  = in_sizes[5];          // 1024
    const int IN  = nb * 4;               // 4096
    const int OUT = in_sizes[2];          // 4096
    const int B   = in_sizes[0] / IN;     // 1024

    _Float16* xq = (_Float16*)d_ws;                       // B*IN f16  (8 MB)
    _Float16* wq = xq + (size_t)B * IN;                   // OUT*IN f16 (32 MB)

    {
        int total = B * nb;
        fq_x_kernel<<<(total + 255) / 256, 256, 0, stream>>>(
            x, a_scales, a_zeros, xq, total, nb);
    }
    {
        int total = OUT * nb;
        fq_w_kernel<<<(total + 255) / 256, 256, 0, stream>>>(
            weight, w_scales, w_zeros, wq, total);
    }
    {
        dim3 grid(OUT / 128, B / 128);    // 32 x 8 blocks, 8 waves each
        wmma_gemm_kernel<<<grid, 256, 0, stream>>>(xq, wq, bias, out, IN, OUT);
    }
}
